// Generator_32461362823919
// MI455X (gfx1250) — matrix operand order repaired and checked by reading, not compiled, not run on hardware
//
#include <hip/hip_runtime.h>
#include <hip/hip_bf16.h>

// ---------------- problem constants ----------------
#define B_     256
#define NOISE_ 64
#define HID_   256
#define EMB_   128
#define VOC_   32000
#define T_     16
#define G4_    1024          // 4*HID
#define EPS_   1e-10f

// logits-kernel tiling: 2000 vocab tiles of 16, 25 blocks * 8 waves = 200 chunks
#define NBLK_V      25
#define NCHUNK      200
#define VT_PER_WAVE 10       // 200*10*16 == 32000

typedef _Float16 h8   __attribute__((ext_vector_type(8)));
typedef _Float16 v16h __attribute__((ext_vector_type(16)));
typedef float    v8f  __attribute__((ext_vector_type(8)));

#define MK16(lo, hi) __builtin_shufflevector(lo, hi, 0,1,2,3,4,5,6,7,8,9,10,11,12,13,14,15)

static __device__ __forceinline__ int imin_(int a, int b) { return a < b ? a : b; }

// merge two online-softmax partial states (max m, sum s of exp(z-m), argmax ix)
static __device__ __forceinline__ void smax_merge(float& m, float& s, int& ix,
                                                  float m2, float s2, int i2) {
  float M  = fmaxf(m, m2);
  float sN = s * __expf(m - M) + s2 * __expf(m2 - M);
  int   iN = (m > m2) ? ix : ((m2 > m) ? i2 : imin_(ix, i2));
  m = M; s = sN; ix = iN;
}

// ---------------- utility kernels ----------------
__global__ void k_zero(float4* __restrict__ out, size_t n4) {
  size_t stride = (size_t)gridDim.x * blockDim.x;
  for (size_t i = (size_t)blockIdx.x * blockDim.x + threadIdx.x; i < n4; i += stride)
    out[i] = make_float4(0.f, 0.f, 0.f, 0.f);
}

__global__ void k_cvt(const float* __restrict__ outW, const float* __restrict__ Wih,
                      const float* __restrict__ Whh,
                      _Float16* __restrict__ outW16, _Float16* __restrict__ Wih16,
                      _Float16* __restrict__ Whh16) {
  const int n0 = VOC_ * HID_, n1 = G4_ * EMB_, n2 = G4_ * HID_;
  int stride = gridDim.x * blockDim.x;
  for (int i = blockIdx.x * blockDim.x + threadIdx.x; i < n0 + n1 + n2; i += stride) {
    if (i < n0)            outW16[i]           = (_Float16)outW[i];
    else if (i < n0 + n1)  Wih16[i - n0]       = (_Float16)Wih[i - n0];
    else                   Whh16[i - n0 - n1]  = (_Float16)Whh[i - n0 - n1];
  }
}

// h0 = tanh(noise @ fc_W.T + fc_b); c=0; x = embed_W[PAD=0]
__global__ void k_init(const float* __restrict__ noise, const float* __restrict__ fc_W,
                       const float* __restrict__ fc_b, const float* __restrict__ embed_W,
                       float* __restrict__ c32, _Float16* __restrict__ h16,
                       _Float16* __restrict__ x16) {
  int idx = blockIdx.x * blockDim.x + threadIdx.x;   // B*HID
  if (idx >= B_ * HID_) return;
  int b = idx >> 8, j = idx & 255;
  const float* nz = noise + b * NOISE_;
  const float* w  = fc_W + j * NOISE_;
  float acc = fc_b[j];
#pragma unroll 8
  for (int k = 0; k < NOISE_; ++k) acc += nz[k] * w[k];
  h16[idx] = (_Float16)tanhf(acc);
  c32[idx] = 0.f;
  if (j < EMB_) x16[b * EMB_ + j] = (_Float16)embed_W[j];
}

// ---------------- gates GEMM: gates = x@Wih.T + h@Whh.T + biases ----------------
// 1024 wave-tiles: 16 batch tiles x 64 gate tiles, one 16x16 tile per wave.
__global__ void __launch_bounds__(256)
k_gates(const _Float16* __restrict__ x16, const _Float16* __restrict__ h16,
        const _Float16* __restrict__ Wih16, const _Float16* __restrict__ Whh16,
        const float* __restrict__ b_ih, const float* __restrict__ b_hh,
        float* __restrict__ gates) {
  int wave = threadIdx.x >> 5, lane = threadIdx.x & 31;
  int tile = blockIdx.x * 8 + wave;          // 0..1023
  int b0 = (tile >> 6) * 16;                 // batch tile
  int n0 = (tile & 63) * 16;                 // gate tile
  int l  = lane & 15;
  int hi = lane >> 4;

  v8f acc = {};
  // x (K=128) * W_ih
  {
    const _Float16* ap = x16   + (size_t)(b0 + l) * EMB_;
    const _Float16* bp = Wih16 + (size_t)(n0 + l) * EMB_;
#pragma unroll
    for (int k0 = 0; k0 < EMB_; k0 += 32) {
      h8 alo = *(const h8*)(ap + k0 + (hi ? 8 : 0));
      h8 ahi = *(const h8*)(ap + k0 + 16 + (hi ? 8 : 0));
      h8 blo = *(const h8*)(bp + k0 + (hi ? 8 : 0));
      h8 bhi = *(const h8*)(bp + k0 + 16 + (hi ? 8 : 0));
      v16h A = MK16(alo, ahi), Bf = MK16(blo, bhi);
      acc = __builtin_amdgcn_wmma_f32_16x16x32_f16(false, A, false, Bf, (short)0, acc,
                                                   false, false);
    }
  }
  // h (K=256) * W_hh
  {
    const _Float16* ap = h16   + (size_t)(b0 + l) * HID_;
    const _Float16* bp = Whh16 + (size_t)(n0 + l) * HID_;
#pragma unroll
    for (int k0 = 0; k0 < HID_; k0 += 32) {
      h8 alo = *(const h8*)(ap + k0 + (hi ? 8 : 0));
      h8 ahi = *(const h8*)(ap + k0 + 16 + (hi ? 8 : 0));
      h8 blo = *(const h8*)(bp + k0 + (hi ? 8 : 0));
      h8 bhi = *(const h8*)(bp + k0 + 16 + (hi ? 8 : 0));
      v16h A = MK16(alo, ahi), Bf = MK16(blo, bhi);
      acc = __builtin_amdgcn_wmma_f32_16x16x32_f16(false, A, false, Bf, (short)0, acc,
                                                   false, false);
    }
  }
  int col  = n0 + l;
  float bias = b_ih[col] + b_hh[col];
  int row0 = b0 + (hi ? 8 : 0);
#pragma unroll
  for (int r = 0; r < 8; ++r)
    gates[(size_t)(row0 + r) * G4_ + col] = acc[r] + bias;
}

// ---------------- LSTM pointwise cell ----------------
__global__ void k_cell(const float* __restrict__ gates, float* __restrict__ c32,
                       _Float16* __restrict__ h16) {
  int idx = blockIdx.x * blockDim.x + threadIdx.x;   // B*HID
  if (idx >= B_ * HID_) return;
  int b = idx >> 8, j = idx & 255;
  const float* g = gates + (size_t)b * G4_;
  float ig = 1.f / (1.f + __expf(-g[j]));
  float fg = 1.f / (1.f + __expf(-g[j + 256]));
  float gg = tanhf(g[j + 512]);
  float og = 1.f / (1.f + __expf(-g[j + 768]));
  float c = fg * c32[idx] + ig * gg;
  c32[idx] = c;
  h16[idx] = (_Float16)(og * tanhf(c));
}

// ---------------- logits GEMM + fused gumbel + online max/sum/argmax ----------------
__global__ void __launch_bounds__(256)
k_logits(const _Float16* __restrict__ h16, const _Float16* __restrict__ outW16,
         const float* __restrict__ out_b, const float* __restrict__ gum,  // + t*B*V
         float* __restrict__ pm, float* __restrict__ ps, int* __restrict__ pidx) {
  int wave = threadIdx.x >> 5, lane = threadIdx.x & 31;
  int l = lane & 15, hi = lane >> 4;
  int b0 = blockIdx.y * 16;
  int chunk = blockIdx.x * 8 + wave;         // 0..199

  // preload all A (h) fragments for this batch tile: K=256 -> 8 fragments
  v16h afr[8];
  {
    const _Float16* ap = h16 + (size_t)(b0 + l) * HID_;
#pragma unroll
    for (int kb = 0; kb < 8; ++kb) {
      int k0 = kb * 32;
      h8 alo = *(const h8*)(ap + k0 + (hi ? 8 : 0));
      h8 ahi = *(const h8*)(ap + k0 + 16 + (hi ? 8 : 0));
      afr[kb] = MK16(alo, ahi);
    }
  }

  float m[8], s[8]; int ix[8];
#pragma unroll
  for (int r = 0; r < 8; ++r) { m[r] = -INFINITY; s[r] = 0.f; ix[r] = 0; }
  int row0 = b0 + (hi ? 8 : 0);

  for (int tI = 0; tI < VT_PER_WAVE; ++tI) {
    int v0 = (chunk * VT_PER_WAVE + tI) * 16;
    v8f acc = {};
    const _Float16* bp = outW16 + (size_t)(v0 + l) * HID_ + (hi ? 8 : 0);
#pragma unroll
    for (int kb = 0; kb < 8; ++kb) {
      h8 blo = *(const h8*)(bp + kb * 32);
      h8 bhi = *(const h8*)(bp + kb * 32 + 16);
      v16h Bf = MK16(blo, bhi);
      acc = __builtin_amdgcn_wmma_f32_16x16x32_f16(false, afr[kb], false, Bf, (short)0,
                                                   acc, false, false);
    }
    int col = v0 + l;
    float ob = out_b[col];
    const float* up = gum + (size_t)row0 * VOC_ + col;
#pragma unroll
    for (int r = 0; r < 8; ++r) {
      float u = up[(size_t)r * VOC_];
      float g = -__logf(-__logf(u + EPS_) + EPS_);
      float z = acc[r] + ob + g;               // TAU == 1
      if (z > m[r]) { s[r] = s[r] * __expf(m[r] - z) + 1.0f; m[r] = z; ix[r] = col; }
      else          { s[r] += __expf(z - m[r]); }
    }
  }

  // reduce across the 16 lanes of each half (xor masks keep halves separate)
#pragma unroll
  for (int off = 1; off <= 8; off <<= 1) {
#pragma unroll
    for (int r = 0; r < 8; ++r) {
      float m2 = __shfl_xor(m[r], off, 32);
      float s2 = __shfl_xor(s[r], off, 32);
      int   i2 = __shfl_xor(ix[r], off, 32);
      smax_merge(m[r], s[r], ix[r], m2, s2, i2);
    }
  }
  if (l == 0) {   // lanes 0 (rows b0..b0+7) and 16 (rows b0+8..b0+15)
#pragma unroll
    for (int r = 0; r < 8; ++r) {
      size_t o = (size_t)(row0 + r) * NCHUNK + chunk;
      pm[o] = m[r]; ps[o] = s[r]; pidx[o] = ix[r];
    }
  }
}

// ---------------- finalize: global argmax/softmax-denom, scatter token, gather embed ----
__global__ void k_combine(const float* __restrict__ pm, const float* __restrict__ ps,
                          const int* __restrict__ pidx, const float* __restrict__ embed_W,
                          float* __restrict__ out, int t, _Float16* __restrict__ x16) {
  int b = blockIdx.x, lane = threadIdx.x;    // one wave32 per batch row
  float m = -INFINITY, s = 0.f; int ix = 0x7fffffff;
  for (int c = lane; c < NCHUNK; c += 32) {
    size_t o = (size_t)b * NCHUNK + c;
    smax_merge(m, s, ix, pm[o], ps[o], pidx[o]);
  }
#pragma unroll
  for (int off = 16; off >= 1; off >>= 1) {
    float m2 = __shfl_xor(m, off, 32);
    float s2 = __shfl_xor(s, off, 32);
    int   i2 = __shfl_xor(ix, off, 32);
    smax_merge(m, s, ix, m2, s2, i2);
  }
  if (lane == 0) {
    float p = 1.0f / s;                               // softmax value at argmax
    out[(size_t)b * T_ * VOC_ + (size_t)t * VOC_ + ix] = (1.0f - p) + p;  // straight-through
  }
  for (int e = lane; e < EMB_; e += 32)               // x_next = embed_W[idx]
    x16[b * EMB_ + e] = (_Float16)embed_W[(size_t)ix * EMB_ + e];
}

// ---------------- host launch ----------------
extern "C" void kernel_launch(void* const* d_in, const int* in_sizes, int n_in,
                              void* d_out, int out_size, void* d_ws, size_t ws_size,
                              hipStream_t stream) {
  (void)in_sizes; (void)n_in; (void)out_size; (void)ws_size;
  const float* noise   = (const float*)d_in[0];
  const float* gumbel  = (const float*)d_in[1];
  const float* fc_W    = (const float*)d_in[2];
  const float* fc_b    = (const float*)d_in[3];
  const float* embed_W = (const float*)d_in[4];
  const float* W_ih    = (const float*)d_in[5];
  const float* W_hh    = (const float*)d_in[6];
  const float* b_ih    = (const float*)d_in[7];
  const float* b_hh    = (const float*)d_in[8];
  const float* out_W   = (const float*)d_in[9];
  const float* out_b   = (const float*)d_in[10];
  float* out = (float*)d_out;
  char*  ws  = (char*)d_ws;

  // workspace layout (~18.5 MB)
  size_t off = 0;
  float*    c32    = (float*)(ws + off);    off += (size_t)B_ * HID_ * 4;        // 256K
  float*    gates  = (float*)(ws + off);    off += (size_t)B_ * G4_ * 4;         // 1M
  _Float16* h16    = (_Float16*)(ws + off); off += (size_t)B_ * HID_ * 2;        // 128K
  _Float16* x16    = (_Float16*)(ws + off); off += (size_t)B_ * EMB_ * 2;        // 64K
  _Float16* Wih16  = (_Float16*)(ws + off); off += (size_t)G4_ * EMB_ * 2;       // 256K
  _Float16* Whh16  = (_Float16*)(ws + off); off += (size_t)G4_ * HID_ * 2;       // 512K
  _Float16* outW16 = (_Float16*)(ws + off); off += (size_t)VOC_ * HID_ * 2;      // 16M
  float*    pm     = (float*)(ws + off);    off += (size_t)B_ * NCHUNK * 4;
  float*    ps     = (float*)(ws + off);    off += (size_t)B_ * NCHUNK * 4;
  int*      pidx   = (int*)(ws + off);      off += (size_t)B_ * NCHUNK * 4;

  // one-time per launch
  k_zero<<<4096, 256, 0, stream>>>((float4*)out, (size_t)B_ * T_ * VOC_ / 4);
  k_cvt <<<8192, 256, 0, stream>>>(out_W, W_ih, W_hh, outW16, Wih16, Whh16);
  k_init<<<(B_ * HID_) / 256, 256, 0, stream>>>(noise, fc_W, fc_b, embed_W, c32, h16, x16);

  for (int t = 0; t < T_; ++t) {
    k_gates<<<128, 256, 0, stream>>>(x16, h16, Wih16, Whh16, b_ih, b_hh, gates);
    k_cell <<<(B_ * HID_) / 256, 256, 0, stream>>>(gates, c32, h16);
    k_logits<<<dim3(NBLK_V, 16), 256, 0, stream>>>(h16, outW16, out_b,
                                                   gumbel + (size_t)t * B_ * VOC_,
                                                   pm, ps, pidx);
    k_combine<<<B_, 32, 0, stream>>>(pm, ps, pidx, embed_W, out, t, x16);
  }
}
